// RobotEncoder_83777632076511
// MI455X (gfx1250) — matrix operand order repaired and checked by reading, not compiled, not run on hardware
//
#include <hip/hip_runtime.h>
#include <stdint.h>
#include <math.h>

#define BATCH 8192
#define NEXP  8
#define DDIM  16
#define FBAND 16
#define HDIM  512
#define GDIM  16
#define TWO_PI 6.28318530717958647692f

typedef __attribute__((ext_vector_type(16))) __bf16          v16bf;
typedef __attribute__((ext_vector_type(16))) unsigned short  v16u16;
typedef __attribute__((ext_vector_type(8)))  float           v8f;
typedef __attribute__((ext_vector_type(4)))  unsigned int    u32x4;
typedef __attribute__((ext_vector_type(8)))  int             i32x8;
typedef __attribute__((ext_vector_type(4)))  int             i32x4;

#if __has_builtin(__builtin_amdgcn_tensor_load_to_lds)
#define HAVE_TDM 1
#else
#define HAVE_TDM 0
#endif

// ---------------------------------------------------------------------------
// TDM: stage a [32 x 512] tile of 2-byte elements (row stride 512 elems,
// contiguous in both global and LDS) via the Tensor Data Mover.
// D# built per CDNA5 ISA ch.8: group0 = {count, lds_addr, global_addr, type=2},
// group1 = {data_size=2B, tensor_dim0=512, tensor_dim1=32, tile_dim0=512,
//           tile_dim1=32, tensor_dim0_stride=512}.
// ---------------------------------------------------------------------------
__device__ __forceinline__ void tdm_issue_rows32(void* lds_dst, const unsigned short* gsrc) {
#if HAVE_TDM
  unsigned long long ga = (unsigned long long)(uintptr_t)gsrc;
  unsigned lds_off = (unsigned)(uintptr_t)lds_dst;   // generic LDS addr[31:0] == LDS byte offset
  u32x4 g0;
  g0.x = 1u;                                          // count=1, user descriptor
  g0.y = lds_off;                                     // lds_addr
  g0.z = (unsigned)(ga & 0xffffffffull);              // global_addr[31:0]
  g0.w = (unsigned)((ga >> 32) & 0x1ffffffull) | (2u << 30);  // global_addr[56:32] | type=2
  i32x8 g1 = { (int)(1u << 16),      // data_size = 1 -> 2 bytes
               (int)(512u << 16),    // tensor_dim0[15:0] in bits[63:48]
               (int)(32u << 16),     // tensor_dim0 hi=0 | tensor_dim1 lo = 32
               (int)(512u << 16),    // tensor_dim1 hi=0 | tile_dim0 = 512
               32,                   // tile_dim1 = 32, tile_dim2 = 0
               512,                  // tensor_dim0_stride[31:0]
               0, 0 };
  i32x4 gz = {0, 0, 0, 0};
#if defined(__clang_major__) && (__clang_major__ >= 23)
  i32x8 gz8 = {0, 0, 0, 0, 0, 0, 0, 0};
  __builtin_amdgcn_tensor_load_to_lds(g0, g1, gz, gz, gz8, 0);
#else
  __builtin_amdgcn_tensor_load_to_lds(g0, g1, gz, gz, 0);
#endif
#else
  (void)lds_dst; (void)gsrc;
#endif
}

__device__ __forceinline__ void tdm_wait_all() {
#if HAVE_TDM && __has_builtin(__builtin_amdgcn_s_wait_tensorcnt)
  __builtin_amdgcn_s_wait_tensorcnt(0);
#endif
}

__device__ __forceinline__ float gelu_exact(float x) {
  return 0.5f * x * (1.0f + erff(x * 0.70710678118654752f));
}

// ---------------------------------------------------------------------------
// fp32 -> bf16 weight repack kernels (into workspace)
// ---------------------------------------------------------------------------
__global__ void conv_w1_kernel(const float* __restrict__ w1, unsigned short* __restrict__ w1b) {
  int idx = blockIdx.x * blockDim.x + threadIdx.x;   // < E*D*32*H
  int h  = idx & (HDIM - 1);
  int f  = (idx >> 9) & 31;
  int ed = idx >> 14;
  float v = w1[(ed * 33 + f) * HDIM + h];            // drop f==32 row (handled in VALU)
  w1b[idx] = __builtin_bit_cast(unsigned short, (__bf16)v);
}

__global__ void conv_wo_kernel(const float* __restrict__ wo, unsigned short* __restrict__ wob) {
  int idx = blockIdx.x * blockDim.x + threadIdx.x;   // < E*H*H
  wob[idx] = __builtin_bit_cast(unsigned short, (__bf16)wo[idx]);
}

// ---------------------------------------------------------------------------
// Gate: per-task linear -> softmax -> top-2 -> renormalize
// ---------------------------------------------------------------------------
__global__ void gate_kernel(const float* __restrict__ gin, const int* __restrict__ task_p,
                            const float* __restrict__ gw, const float* __restrict__ gb,
                            float* __restrict__ gates, float* __restrict__ loss_out) {
  int b = blockIdx.x * blockDim.x + threadIdx.x;
  if (b == 0) loss_out[0] = 0.0f;
  if (b >= BATCH) return;
  int task = task_p[0];
  const float* w  = gw + (size_t)task * GDIM * NEXP;
  const float* bb = gb + (size_t)task * NEXP;
  float x[GDIM];
#pragma unroll
  for (int i = 0; i < GDIM; ++i) x[i] = gin[b * GDIM + i];
  float lg[NEXP];
#pragma unroll
  for (int e = 0; e < NEXP; ++e) {
    float s = bb[e];
#pragma unroll
    for (int i = 0; i < GDIM; ++i) s += x[i] * w[i * NEXP + e];
    lg[e] = s;
  }
  float mx = lg[0];
#pragma unroll
  for (int e = 1; e < NEXP; ++e) mx = fmaxf(mx, lg[e]);
  float pe[NEXP];
#pragma unroll
  for (int e = 0; e < NEXP; ++e) pe[e] = __expf(lg[e] - mx);
  // top-2 of pe == top-2 of softmax; renorm ratio is softmax-denominator invariant
  int i1 = 0; float v1 = pe[0];
#pragma unroll
  for (int e = 1; e < NEXP; ++e) if (pe[e] > v1) { v1 = pe[e]; i1 = e; }
  int i2 = -1; float v2 = -1.0f;
#pragma unroll
  for (int e = 0; e < NEXP; ++e) if (e != i1 && pe[e] > v2) { v2 = pe[e]; i2 = e; }
  float inv = 1.0f / (v1 + v2);
#pragma unroll
  for (int e = 0; e < NEXP; ++e)
    gates[b * NEXP + e] = (e == i1) ? v1 * inv : ((e == i2) ? v2 * inv : 0.0f);
}

// ---------------------------------------------------------------------------
// Fused MoE expert kernel: 16-row tile x 512 cols per workgroup (8 waves).
// GEMM1 (K=32 cos|sin features, bf16 WMMA) + bias + x-rank1 + LN + GELU +
// d-sum, then GEMM2 (K=512, bf16 WMMA) + bo, gate-scaled accumulation.
// ---------------------------------------------------------------------------
__global__ __launch_bounds__(256) void moe_kernel(
    const float* __restrict__ xin,            // [B,16]
    const float* __restrict__ freqs,          // [E,16,16]
    const float* __restrict__ w1,             // fp32 [E,16,33,512] (for f=32 row)
    const float* __restrict__ b1,             // [E,16,512]
    const float* __restrict__ lng,            // [E,16,512]
    const float* __restrict__ lnb,            // [E,16,512]
    const float* __restrict__ bo,             // [E,512]
    const unsigned short* __restrict__ w1b,   // bf16 [E,16,32,512]
    const unsigned short* __restrict__ wob,   // bf16 [E,512,512]
    const float* __restrict__ gates,          // [B,8]
    float* __restrict__ out)                  // [B,512]
{
  __shared__ __align__(16) unsigned short slab[32 * HDIM];     // 32KB: w1 d-slab / wo k-slab
  __shared__ __align__(16) unsigned short hsum_s[16 * HDIM];   // 16KB: bf16 d-summed acts
  __shared__ float xi_s[16 * 16];    // [row][d]
  __shared__ float fr_s[16 * 16];    // [d][f] current expert
  __shared__ float gate_s[16 * NEXP];
  __shared__ float lnS[16];
  __shared__ float lnQ[16];

  const int tid    = threadIdx.x;
  const int wave   = tid >> 5;
  const int lane   = tid & 31;
  const int lane15 = lane & 15;
  const int hi     = lane >> 4;
  const int rowbase = blockIdx.x * 16;

  // 16-bit A-fragment K map (CDNA5 ISA 7.12.2): VGPR j<4 -> K=2j(+8 hi half),
  // j>=4 -> K=16+2(j-4)(+8 hi half); element pair (2j,2j+1) -> (K,K+1).
  int kmap[8];
#pragma unroll
  for (int j = 0; j < 8; ++j) kmap[j] = ((j < 4) ? 2 * j : 16 + 2 * (j - 4)) + hi * 8;

  if (tid < 256) xi_s[tid]  = xin[(rowbase + (tid >> 4)) * DDIM + (tid & 15)];
  if (tid < 128) gate_s[tid] = gates[(rowbase + (tid >> 3)) * NEXP + (tid & 7)];
  if (tid < 16) { lnS[tid] = 0.0f; lnQ[tid] = 0.0f; }

  float go[4][8];
#pragma unroll
  for (int t = 0; t < 4; ++t)
#pragma unroll
    for (int v = 0; v < 8; ++v) go[t][v] = 0.0f;

  for (int e = 0; e < NEXP; ++e) {
    __syncthreads();                                  // previous expert fully done
    if (tid < 256) fr_s[tid] = freqs[e * 256 + tid];

    float hs[4][8];
#pragma unroll
    for (int t = 0; t < 4; ++t)
#pragma unroll
      for (int v = 0; v < 8; ++v) hs[t][v] = 0.0f;

    // ------------------- GEMM1 + LN + GELU, summed over d -------------------
    for (int d = 0; d < DDIM; ++d) {
      __syncthreads();                                // slab free, lnS/lnQ zeroed, fr_s ready
#if HAVE_TDM
      if (wave == 0) tdm_issue_rows32(slab, w1b + ((size_t)(e * DDIM + d) * 32) * HDIM);
#else
      {
        const uint4* s = (const uint4*)(w1b + ((size_t)(e * DDIM + d) * 32) * HDIM);
        uint4* dl = (uint4*)slab;
        for (int i = tid; i < (32 * HDIM) / 8; i += 256) dl[i] = s[i];
      }
#endif
      // A fragment: synthesize cos/sin features directly into WMMA layout
      v16bf afrag;
      {
        float x = xi_s[lane15 * DDIM + d];
#pragma unroll
        for (int j = 0; j < 8; ++j) {
          int K = kmap[j];
          int f = K & 15;
          float a0 = x * fr_s[d * 16 + f] * TWO_PI;
          float a1 = x * fr_s[d * 16 + f + 1] * TWO_PI;
          float s0, s1;
          if (j < 4) { s0 = __cosf(a0); s1 = __cosf(a1); }   // K<16 -> cos
          else       { s0 = __sinf(a0); s1 = __sinf(a1); }   // K>=16 -> sin
          afrag[2 * j]     = (__bf16)s0;
          afrag[2 * j + 1] = (__bf16)s1;
        }
      }
      float xiv[8];
#pragma unroll
      for (int v = 0; v < 8; ++v) xiv[v] = xi_s[(v + hi * 8) * DDIM + d];

#if HAVE_TDM
      if (wave == 0) tdm_wait_all();
#endif
      __syncthreads();                                // slab ready for all waves

      const int base = (e * DDIM + d) * HDIM;
      float hc[4][8];
#pragma unroll
      for (int t = 0; t < 4; ++t) {
        const int n0 = (wave * 4 + t) * 16;
        v16u16 bu;   // n = n0 + lane15, K per kmap
#pragma unroll
        for (int j = 0; j < 8; ++j) {
          bu[2 * j]     = slab[kmap[j] * HDIM + n0 + lane15];
          bu[2 * j + 1] = slab[(kmap[j] + 1) * HDIM + n0 + lane15];
        }
        v16bf bfrag = __builtin_bit_cast(v16bf, bu);
        v8f c = {0.f, 0.f, 0.f, 0.f, 0.f, 0.f, 0.f, 0.f};
        c = __builtin_amdgcn_wmma_f32_16x16x32_bf16(false, afrag, false, bfrag,
                                                    (short)0, c, false, false);
        const int col = n0 + lane15;
        const float b1v = b1[base + col];
        const float w32 = w1[((size_t)(e * DDIM + d) * 33 + 32) * HDIM + col];
#pragma unroll
        for (int v = 0; v < 8; ++v) hc[t][v] = c[v] + b1v + xiv[v] * w32;
      }

      // LayerNorm stats across H=512 (4 tiles in-lane, 16 lanes, 8 waves)
      float ps[8], pq[8];
#pragma unroll
      for (int v = 0; v < 8; ++v) {
        float s = hc[0][v] + hc[1][v] + hc[2][v] + hc[3][v];
        float q = hc[0][v] * hc[0][v] + hc[1][v] * hc[1][v] +
                  hc[2][v] * hc[2][v] + hc[3][v] * hc[3][v];
        ps[v] = s; pq[v] = q;
      }
#pragma unroll
      for (int m = 1; m < 16; m <<= 1) {
#pragma unroll
        for (int v = 0; v < 8; ++v) {
          ps[v] += __shfl_xor(ps[v], m, 32);
          pq[v] += __shfl_xor(pq[v], m, 32);
        }
      }
      if (lane15 == 0) {
#pragma unroll
        for (int v = 0; v < 8; ++v) {
          atomicAdd(&lnS[v + hi * 8], ps[v]);
          atomicAdd(&lnQ[v + hi * 8], pq[v]);
        }
      }
      __syncthreads();

      float mean[8], rstd[8];
#pragma unroll
      for (int v = 0; v < 8; ++v) {
        int r = v + hi * 8;
        float m  = lnS[r] * (1.0f / HDIM);
        float vr = lnQ[r] * (1.0f / HDIM) - m * m;
        mean[v] = m;
        rstd[v] = rsqrtf(vr + 1e-5f);
      }
#pragma unroll
      for (int t = 0; t < 4; ++t) {
        const int col = (wave * 4 + t) * 16 + lane15;
        const float g = lng[base + col];
        const float bb = lnb[base + col];
#pragma unroll
        for (int v = 0; v < 8; ++v) {
          float xn = (hc[t][v] - mean[v]) * rstd[v] * g + bb;
          hs[t][v] += gelu_exact(xn);
        }
      }
      __syncthreads();
      if (tid < 16) { lnS[tid] = 0.0f; lnQ[tid] = 0.0f; }
    } // d

    // d-summed activations -> LDS as bf16 (GEMM2 A operand source)
#pragma unroll
    for (int t = 0; t < 4; ++t) {
      const int col = (wave * 4 + t) * 16 + lane15;
#pragma unroll
      for (int v = 0; v < 8; ++v)
        hsum_s[(v + hi * 8) * HDIM + col] =
            __builtin_bit_cast(unsigned short, (__bf16)hs[t][v]);
    }

    // --------------------------- GEMM2: [16,512]x[512,512] ---------------------------
    v8f oc[4];
#pragma unroll
    for (int t = 0; t < 4; ++t) oc[t] = (v8f){0.f,0.f,0.f,0.f,0.f,0.f,0.f,0.f};

    for (int kk = 0; kk < 16; ++kk) {
      __syncthreads();                                // hsum_s written / slab free
#if HAVE_TDM
      if (wave == 0) tdm_issue_rows32(slab, wob + ((size_t)e * HDIM + kk * 32) * HDIM);
#else
      {
        const uint4* s = (const uint4*)(wob + ((size_t)e * HDIM + kk * 32) * HDIM);
        uint4* dl = (uint4*)slab;
        for (int i = tid; i < (32 * HDIM) / 8; i += 256) dl[i] = s[i];
      }
#endif
      // A fragment from LDS (m = lane15 row, K per kmap)
      v16u16 au;
#pragma unroll
      for (int j = 0; j < 8; ++j) {
        unsigned pr = *(const unsigned*)(hsum_s + lane15 * HDIM + kk * 32 + kmap[j]);
        au[2 * j]     = (unsigned short)(pr & 0xffffu);
        au[2 * j + 1] = (unsigned short)(pr >> 16);
      }
      v16bf afrag = __builtin_bit_cast(v16bf, au);

#if HAVE_TDM
      if (wave == 0) tdm_wait_all();
#endif
      __syncthreads();

#pragma unroll
      for (int t = 0; t < 4; ++t) {
        const int n0 = (wave * 4 + t) * 16;
        v16u16 bu;
#pragma unroll
        for (int j = 0; j < 8; ++j) {
          bu[2 * j]     = slab[kmap[j] * HDIM + n0 + lane15];
          bu[2 * j + 1] = slab[(kmap[j] + 1) * HDIM + n0 + lane15];
        }
        v16bf bfrag = __builtin_bit_cast(v16bf, bu);
        oc[t] = __builtin_amdgcn_wmma_f32_16x16x32_bf16(false, afrag, false, bfrag,
                                                        (short)0, oc[t], false, false);
      }
    } // kk

    // Epilogue: + bo, gate-scaled accumulation
#pragma unroll
    for (int t = 0; t < 4; ++t) {
      const int col = (wave * 4 + t) * 16 + lane15;
      const float bov = bo[e * HDIM + col];
#pragma unroll
      for (int v = 0; v < 8; ++v) {
        const float gval = gate_s[(v + hi * 8) * NEXP + e];
        go[t][v] += gval * (oc[t][v] + bov);
      }
    }
  } // experts

#pragma unroll
  for (int t = 0; t < 4; ++t) {
    const int col = (wave * 4 + t) * 16 + lane15;
#pragma unroll
    for (int v = 0; v < 8; ++v)
      out[(size_t)(rowbase + v + hi * 8) * HDIM + col] = go[t][v];
  }
}

// ---------------------------------------------------------------------------
extern "C" void kernel_launch(void* const* d_in, const int* in_sizes, int n_in,
                              void* d_out, int out_size, void* d_ws, size_t ws_size,
                              hipStream_t stream) {
  const float* gate_input   = (const float*)d_in[0];
  const float* expert_input = (const float*)d_in[1];
  const int*   task_p       = (const int*)d_in[2];
  const float* gate_w       = (const float*)d_in[3];
  const float* gate_b       = (const float*)d_in[4];
  const float* freqs        = (const float*)d_in[5];
  const float* w1           = (const float*)d_in[6];
  const float* b1           = (const float*)d_in[7];
  const float* ln_g         = (const float*)d_in[8];
  const float* ln_b         = (const float*)d_in[9];
  const float* wo           = (const float*)d_in[10];
  const float* bo           = (const float*)d_in[11];
  float* out = (float*)d_out;

  unsigned short* w1b = (unsigned short*)d_ws;                       // E*D*32*H bf16 = 4 MiB
  unsigned short* wob = w1b + (size_t)NEXP * DDIM * 32 * HDIM;       // E*H*H bf16   = 4 MiB
  float* gates = (float*)(wob + (size_t)NEXP * HDIM * HDIM);         // B*E f32      = 256 KiB

  conv_w1_kernel<<<(NEXP * DDIM * 32 * HDIM) / 256, 256, 0, stream>>>(w1, w1b);
  conv_wo_kernel<<<(NEXP * HDIM * HDIM) / 256, 256, 0, stream>>>(wo, wob);
  gate_kernel<<<BATCH / 256, 256, 0, stream>>>(gate_input, task_p, gate_w, gate_b,
                                               gates, out + (size_t)BATCH * HDIM);
  moe_kernel<<<BATCH / 16, 256, 0, stream>>>(expert_input, freqs, w1, b1, ln_g, ln_b,
                                             bo, w1b, wob, gates, out);
}
